// SqueezeExcite_94489280931
// MI455X (gfx1250) — matrix operand order repaired and checked by reading, not compiled, not run on hardware
//
#include <hip/hip_runtime.h>
#include <hip/hip_bf16.h>
#include <math.h>

// ---------------------------------------------------------------------------
// SqueezeExcite-style gated segment op for MI455X (gfx1250, wave32).
//
// Roofline: 3.07 GB of HBM traffic (x read twice + out written once) at
// 23.3 TB/s -> ~132 us floor. The 2-layer MLP is ~4.2 MFLOP -> noise.
// Strategy: streaming kernels for the heavy parts, one-WGP WMMA kernel
// (v_wmma_f32_16x16x32_f16) for the gate MLP.
// ---------------------------------------------------------------------------

typedef __attribute__((ext_vector_type(16))) _Float16 v16h;
typedef __attribute__((ext_vector_type(8)))  float    v8f;

#define CDIM 256   // channels
#define HDIM 64    // hidden
#define SEGS 64    // segments
#define SR_ROWS 128

// ---------------------------------------------------------------------------
__global__ void zero_f32(float* __restrict__ p, int n) {
  int i = blockIdx.x * blockDim.x + threadIdx.x;
  if (i < n) p[i] = 0.0f;
}

// ---------------------------------------------------------------------------
// Segment-sum + counts. batch is sorted, so each block's 128 rows touch very
// few segments: accumulate per-column in a register and flush with one
// atomicAdd per segment run (not per row).
__global__ __launch_bounds__(256)
void seg_reduce(const float* __restrict__ x, const int* __restrict__ batch,
                int N, float* __restrict__ seg_sum, float* __restrict__ counts) {
  __shared__ int sseg[SR_ROWS];
  __shared__ int hist[SEGS];
  const int t  = threadIdx.x;
  const int r0 = blockIdx.x * SR_ROWS;

  if (t < SR_ROWS) {
    int r = r0 + t;
    sseg[t] = (r < N) ? batch[r] : -1;
  }
  if (t < SEGS) hist[t] = 0;
  __syncthreads();
  if (t < SR_ROWS && sseg[t] >= 0) atomicAdd(&hist[sseg[t]], 1);
  __syncthreads();
  if (t < SEGS && hist[t] > 0) atomicAdd(&counts[t], (float)hist[t]);

  // thread t owns column t; rows stream coalesced (256 floats = 1KB/row)
  const int c = t;
  float acc = 0.0f;
  int   cur = sseg[0];
  for (int i = 0; i < SR_ROWS; ++i) {
    int sg = sseg[i];
    if (sg < 0) break;                      // sorted: -1 only at the tail
    if (sg != cur) {
      if (acc != 0.0f) atomicAdd(&seg_sum[cur * CDIM + c], acc);
      acc = 0.0f;
      cur = sg;
    }
    acc += x[(size_t)(r0 + i) * CDIM + c];
  }
  if (cur >= 0 && acc != 0.0f) atomicAdd(&seg_sum[cur * CDIM + c], acc);
}

// ---------------------------------------------------------------------------
// Gate MLP on one WGP: z = seg_sum/count (f32->f16 in LDS),
//   h = relu(z @ W1)  via v_wmma_f32_16x16x32_f16  (M=64,N=64,K=256)
//   s = sigmoid(h @ W2) via v_wmma_f32_16x16x32_f16 (M=64,N=256,K=64)
// 512 threads = 16 waves; GEMM1: 1 tile/wave; GEMM2: 4 tiles/wave.
__global__ __launch_bounds__(512)
void gate_mlp(const float* __restrict__ seg_sum, const float* __restrict__ counts,
              const float* __restrict__ W1, const float* __restrict__ W2,
              float* __restrict__ s_out) {
  __shared__ _Float16 zsh[SEGS][CDIM + 8];   // 33 KB
  __shared__ _Float16 hsh[SEGS][HDIM + 8];   //  9 KB
  const int t = threadIdx.x;

  // phase 0: z = seg_sum / max(count,1), cast to f16 into LDS
  for (int i = t; i < SEGS * CDIM; i += 512) {
    int row = i >> 8, col = i & 255;
    zsh[row][col] = (_Float16)(seg_sum[i] / fmaxf(counts[row], 1.0f));
  }
  __syncthreads();

  const int w     = t >> 5;
  const int lane  = t & 31;
  const int half_ = lane >> 4;   // 0 or 1
  const int lm    = lane & 15;

  // ---- GEMM1: tiles (tm,tn) of 16x16, K loop 8 x 32 --------------------
  {
    const int tm = w >> 2, tn = w & 3;
    v8f acc = {};
    for (int k0 = 0; k0 < CDIM; k0 += 32) {
      v16h a, b;
#pragma unroll
      for (int v = 0; v < 8; ++v) {
        int kb = k0 + ((v < 4) ? 0 : 16) + 2 * (v & 3) + 8 * half_;
        a[2 * v]     = zsh[tm * 16 + lm][kb];
        a[2 * v + 1] = zsh[tm * 16 + lm][kb + 1];
      }
#pragma unroll
      for (int j = 0; j < 8; ++j) {
        int kk = k0 + ((j < 4) ? 0 : 16) + 2 * (j & 3) + 8 * half_;
        int n  = tn * 16 + lm;
        b[2 * j]     = (_Float16)W1[(size_t)kk * HDIM + n];
        b[2 * j + 1] = (_Float16)W1[(size_t)(kk + 1) * HDIM + n];
      }
      acc = __builtin_amdgcn_wmma_f32_16x16x32_f16(
          false, a, false, b, (short)0, acc, false, false);
    }
#pragma unroll
    for (int v = 0; v < 8; ++v) {   // ReLU fused into f16 handoff
      hsh[tm * 16 + v + 8 * half_][tn * 16 + lm] = (_Float16)fmaxf(acc[v], 0.0f);
    }
  }
  __syncthreads();

  // ---- GEMM2: 64 tiles -> 4 per wave, K loop 2 x 32 --------------------
  {
    const int tm  = w & 3;
    const int tnb = (w >> 2) * 4;
    for (int it = 0; it < 4; ++it) {
      const int tn = tnb + it;
      v8f acc = {};
#pragma unroll
      for (int k0 = 0; k0 < HDIM; k0 += 32) {
        v16h a, b;
#pragma unroll
        for (int v = 0; v < 8; ++v) {
          int kb = k0 + ((v < 4) ? 0 : 16) + 2 * (v & 3) + 8 * half_;
          a[2 * v]     = hsh[tm * 16 + lm][kb];
          a[2 * v + 1] = hsh[tm * 16 + lm][kb + 1];
        }
#pragma unroll
        for (int j = 0; j < 8; ++j) {
          int kk = k0 + ((j < 4) ? 0 : 16) + 2 * (j & 3) + 8 * half_;
          int n  = tn * 16 + lm;
          b[2 * j]     = (_Float16)W2[(size_t)kk * CDIM + n];
          b[2 * j + 1] = (_Float16)W2[(size_t)(kk + 1) * CDIM + n];
        }
        acc = __builtin_amdgcn_wmma_f32_16x16x32_f16(
            false, a, false, b, (short)0, acc, false, false);
      }
#pragma unroll
      for (int v = 0; v < 8; ++v) {  // sigmoid fused into the store of s
        float d = acc[v];
        s_out[(size_t)(tm * 16 + v + 8 * half_) * CDIM + tn * 16 + lm] =
            1.0f / (1.0f + __expf(-d));
      }
    }
  }
}

// ---------------------------------------------------------------------------
// out = x * s[batch] : pure streaming, float4 (B128) loads/stores.
// 256 threads handle 32 rows/block; s rows (64 KB total) stay hot in L2.
__global__ __launch_bounds__(256)
void apply_gate(const float* __restrict__ x, const int* __restrict__ batch,
                const float* __restrict__ s, float* __restrict__ out, int N) {
  __shared__ int segs[32];
  const int t  = threadIdx.x;
  const int r0 = blockIdx.x * 32;
  if (t < 32) {
    int r = r0 + t;
    segs[t] = (r < N) ? batch[r] : 0;
  }
  __syncthreads();

  const float4* __restrict__ x4 = (const float4*)x;
  const float4* __restrict__ s4 = (const float4*)s;
  float4* __restrict__ o4       = (float4*)out;
  const int col  = t & 63;    // 64 float4 per 256-float row
  const int rsub = t >> 6;    // 0..3 (uniform within a wave32)

  for (int j = 0; j < 8; ++j) {
    int rl = j * 4 + rsub;
    int r  = r0 + rl;
    if (r < N) {
      int seg = segs[rl];
      size_t xb = (size_t)r * 64 + col;
      size_t sb = (size_t)seg * 64 + col;
      float4 xv = x4[xb];
      float4 sv = s4[sb];
      float4 ov;
      ov.x = xv.x * sv.x;
      ov.y = xv.y * sv.y;
      ov.z = xv.z * sv.z;
      ov.w = xv.w * sv.w;
      o4[xb] = ov;
    }
  }
}

// ---------------------------------------------------------------------------
extern "C" void kernel_launch(void* const* d_in, const int* in_sizes, int n_in,
                              void* d_out, int out_size, void* d_ws, size_t ws_size,
                              hipStream_t stream) {
  const float* x     = (const float*)d_in[0];
  const int*   batch = (const int*)d_in[1];
  const float* W1    = (const float*)d_in[2];
  const float* W2    = (const float*)d_in[3];
  float*       out   = (float*)d_out;
  const int    N     = in_sizes[0] / CDIM;

  // workspace layout (all f32): seg_sum[64*256] | counts[64] | s[64*256]
  float* seg_sum = (float*)d_ws;
  float* counts  = seg_sum + SEGS * CDIM;
  float* s       = counts + SEGS;

  const int nz = SEGS * CDIM + SEGS;  // seg_sum + counts (contiguous)
  zero_f32<<<(nz + 255) / 256, 256, 0, stream>>>(seg_sum, nz);
  seg_reduce<<<(N + SR_ROWS - 1) / SR_ROWS, 256, 0, stream>>>(x, batch, N,
                                                              seg_sum, counts);
  gate_mlp<<<1, 512, 0, stream>>>(seg_sum, counts, W1, W2, s);
  apply_gate<<<(N + 31) / 32, 256, 0, stream>>>(x, batch, s, out, N);
}
